// Force_77979426226452
// MI455X (gfx1250) — matrix operand order repaired and checked by reading, not compiled, not run on hardware
//
#include <hip/hip_runtime.h>
#include <hip/hip_bf16.h>
#include <math.h>

typedef _Float16 h16;
typedef __attribute__((ext_vector_type(16))) _Float16 v16h;
typedef __attribute__((ext_vector_type(8)))  float    v8f;

#define N_NODE 1024
#define N_EDGE 2048
#define VD 128
#define ED 64
#define QD 3
#define H 128
#define M_MOL 32
#define ESPV 1e-6f

// Intra-wave LDS ordering: LDS pipe is in-order per wave (ISA 7.3), lanes are
// lockstep; we only need to stop the compiler from reordering/caching.
#define LDS_FENCE() asm volatile("" ::: "memory")

// ---- WMMA operand assembly (ISA 7.12.2, wave32) ----
// A (16x32 f16): lane L holds row m=L&15; half t -> K=16*hi+t, hi=L>>4.
__device__ __forceinline__ v16h wmma_load_A(const h16* buf, int stride, int kb, int lane) {
  int m = lane & 15, hi = lane >> 4;
  const h16* row = buf + m * stride + kb;
  v16h a;
#pragma unroll
  for (int t = 0; t < 16; ++t) a[t] = row[16*hi + t];
  return a;
}

// B (32x16 f16): lane L holds column n=L&15; half t -> K=16*hi+t.
// WT is the transposed weight, row-major [ncols_out][K].
__device__ __forceinline__ v16h wmma_load_B(const h16* WT, int K, int ntile, int kb, int lane) {
  int n = lane & 15, hi = lane >> 4;
  const h16* p = WT + (size_t)(ntile*16 + n) * K + kb + 16*hi;
  v16h b;
#pragma unroll
  for (int t = 0; t < 16; ++t) b[t] = p[t];
  return b;
}

// B for an H->1 head padded to 16 output columns: column 0 = w, rest 0.
__device__ __forceinline__ v16h wmma_load_Bvec(const h16* w, int kb, int lane) {
  int n = lane & 15, hi = lane >> 4;
  v16h b;
#pragma unroll
  for (int t = 0; t < 16; ++t) b[t] = (n == 0) ? w[kb + 16*hi + t] : (h16)0.0f;
  return b;
}

__device__ __forceinline__ v8f wmma16(v16h a, v16h b, v8f c) {
  return __builtin_amdgcn_wmma_f32_16x16x32_f16(false, a, false, b, (short)0, c, false, false);
}

// ---------------- prep kernels ----------------
__global__ void k_zero(float* p, int n) {
  int i = blockIdx.x * blockDim.x + threadIdx.x;
  if (i < n) p[i] = 0.0f;
}

// one-hot [R][C] (row-major) -> out[c] = r for the nonzero entry of column c
__global__ void k_decode(const float* mat, int* out, int c_log2, int total) {
  int i = blockIdx.x * blockDim.x + threadIdx.x;
  if (i < total && mat[i] != 0.0f) {
    int c = i & ((1 << c_log2) - 1);
    out[c] = i >> c_log2;
  }
}

__global__ void k_group(const int* mol, int* glist, int* gcnt) {
  int mm = threadIdx.x;
  if (mm >= M_MOL) return;
  int cnt = 0;
  for (int n = 0; n < N_NODE; ++n)
    if (mol[n] == mm) glist[mm * N_NODE + cnt++] = n;
  gcnt[mm] = cnt;
}

// src f32 [K][C] -> dst f16 [C][K]  (C = 1<<c_log2)
__global__ void k_transpose16(const float* src, h16* dst, int K, int c_log2) {
  int i = blockIdx.x * blockDim.x + threadIdx.x;
  int total = K << c_log2;
  if (i < total) {
    int k = i >> c_log2;
    int c = i & ((1 << c_log2) - 1);
    dst[c * K + k] = (h16)src[i];
  }
}

__global__ void k_cvt16(const float* src, h16* dst, int n) {
  int i = blockIdx.x * blockDim.x + threadIdx.x;
  if (i < n) dst[i] = (h16)src[i];
}

// a[n][h] = sum_k v[n][k] * frW1[(1+k)][h]   (fp32, tiny GEMM)
__global__ void k_prec_a(const float* v, const float* frW1, float* a) {
  int n = blockIdx.x;
  int h = threadIdx.x;
  float s = 0.0f;
  for (int k = 0; k < VD; ++k)
    s += v[n * VD + k] * frW1[(1 + k) * H + h];
  a[n * H + h] = s;
}

// ---------------- bond forces ----------------
// 4096 edge-directions, 16 per wave-tile; 64 blocks x 4 waves = 256 tiles.
__global__ void __launch_bounds__(128)
k_bond(const float* v, const float* e, const float* q,
       const int* srcI, const int* dstI,
       const h16* W1T, const float* b1,
       const h16* W2T, const float* b2,
       const h16* W3h, const float* b3,
       float* f) {
  __shared__ __align__(16) h16 xbuf_s[4][16 * 192];   // x tile, reused for h2
  __shared__ __align__(16) h16 h1_s[4][16 * 128];
  __shared__ float val_s[4][16];
  __shared__ int un_s[4][16], wn_s[4][16], jj_s[4][16];

  int wave = threadIdx.x >> 5;
  int lane = threadIdx.x & 31;
  int n_lane = lane & 15, hi = lane >> 4;
  int tile = blockIdx.x * 4 + wave;

  h16* xb = xbuf_s[wave];
  h16* h1 = h1_s[wave];
  float* val16 = val_s[wave];
  int* un = un_s[wave]; int* wn = wn_s[wave]; int* jj = jj_s[wave];

  int u_reg = 0, w_reg = 0;
  if (lane < 16) {
    int j = tile * 16 + lane;                    // direction in [0, 2E)
    int j2 = (j < N_EDGE) ? j : j - N_EDGE;
    int uu = (j < N_EDGE) ? srcI[j2] : dstI[j2];
    int ww = (j < N_EDGE) ? dstI[j2] : srcI[j2];
    un[lane] = uu; wn[lane] = ww; jj[lane] = j2;
    u_reg = uu; w_reg = ww;
  }
  LDS_FENCE();

  // x tile: [v_u + v_w (128), e_j (64)] per direction, f16
  for (int idx = lane; idx < 16 * 192; idx += 32) {
    int p = idx / 192;
    int c = idx - p * 192;
    float xv = (c < VD) ? (v[un[p] * VD + c] + v[wn[p] * VD + c])
                        : e[jj[p] * ED + (c - VD)];
    xb[idx] = (h16)xv;
  }
  LDS_FENCE();

  // layer1: h1 = tanh(x @ W1 + b1)   K=192
  for (int nt = 0; nt < 8; ++nt) {
    v8f acc = {};
#pragma unroll
    for (int ks = 0; ks < 6; ++ks) {
      v16h A = wmma_load_A(xb, 192, ks * 32, lane);
      v16h B = wmma_load_B(W1T, 192, nt, ks * 32, lane);
      acc = wmma16(A, B, acc);
    }
    float bias = b1[nt * 16 + n_lane];
#pragma unroll
    for (int r = 0; r < 8; ++r)
      h1[(r + 8 * hi) * 128 + nt * 16 + n_lane] = (h16)tanhf(acc[r] + bias);
  }
  LDS_FENCE();

  // layer2: h2 = tanh(h1 @ W2 + b2) -> reuse xb as [16][128]
  h16* h2 = xb;
  for (int nt = 0; nt < 8; ++nt) {
    v8f acc = {};
#pragma unroll
    for (int ks = 0; ks < 4; ++ks) {
      v16h A = wmma_load_A(h1, 128, ks * 32, lane);
      v16h B = wmma_load_B(W2T, 128, nt, ks * 32, lane);
      acc = wmma16(A, B, acc);
    }
    float bias = b2[nt * 16 + n_lane];
#pragma unroll
    for (int r = 0; r < 8; ++r)
      h2[(r + 8 * hi) * 128 + nt * 16 + n_lane] = (h16)tanhf(acc[r] + bias);
  }
  LDS_FENCE();

  // layer3: val = h2 @ W3 + b3 (W3 padded to 16 cols, col 0 live)
  {
    v8f acc = {};
#pragma unroll
    for (int ks = 0; ks < 4; ++ks) {
      v16h A = wmma_load_A(h2, 128, ks * 32, lane);
      v16h B = wmma_load_Bvec(W3h, ks * 32, lane);
      acc = wmma16(A, B, acc);
    }
    if (n_lane == 0) {
      float bias = b3[0];
#pragma unroll
      for (int r = 0; r < 8; ++r) val16[r + 8 * hi] = acc[r] + bias;
    }
  }
  LDS_FENCE();

  if (lane < 16) {
    float dq0 = q[u_reg * QD + 0] - q[w_reg * QD + 0];
    float dq1 = q[u_reg * QD + 1] - q[w_reg * QD + 1];
    float dq2 = q[u_reg * QD + 2] - q[w_reg * QD + 2];
    float inv = 1.0f / (sqrtf(dq0 * dq0 + dq1 * dq1 + dq2 * dq2) + ESPV);
    float vv = val16[lane];
    atomicAdd(&f[u_reg * QD + 0], dq0 * inv * vv);
    atomicAdd(&f[u_reg * QD + 1], dq1 * inv * vv);
    atomicAdd(&f[u_reg * QD + 2], dq2 * inv * vv);
  }
}

// ---------------- pairwise forces (same-molecule only) ----------------
__global__ void __launch_bounds__(128)
k_pair(const float* q, const float* mvec, const float* a, const float* frW1,
       const h16* W2T, const h16* W3h,
       const int* glist, const int* gcnt,
       float* f) {
  __shared__ __align__(16) h16 h1_s[4][16 * 128];
  __shared__ __align__(16) h16 h2_s[4][16 * 128];
  __shared__ float val_s[4][16];
  __shared__ int jn_s[4][16];
  __shared__ float gk_s[4][16], mk_s[4][16];

  int wave = threadIdx.x >> 5;
  int lane = threadIdx.x & 31;
  int n_lane = lane & 15, hi = lane >> 4;

  int mm = blockIdx.x;
  int gsz = gcnt[mm];
  const int* list = glist + mm * N_NODE;
  int jt_count = (gsz + 15) >> 4;
  int tiles = gsz * jt_count;

  h16* h1 = h1_s[wave];
  h16* h2 = h2_s[wave];
  float* val16 = val_s[wave];
  int* jn = jn_s[wave];
  float* gk = gk_s[wave];
  float* mk = mk_s[wave];

  for (int t = wave; t < tiles; t += 4) {           // wave-uniform loop
    int i_idx = t / jt_count;
    int jt = t - i_idx * jt_count;
    int i = list[i_idx];

    float u0 = 0.f, u1 = 0.f, u2 = 0.f;
    int valid = 0;
    if (lane < 16) {
      int jp = jt * 16 + lane;
      int j = (jp < gsz) ? list[jp] : -1;
      jn[lane] = j;
      valid = (j >= 0);
      float g = 0.0f, msk = 0.0f;
      if (valid) {
        float dq0 = q[i * QD + 0] - q[j * QD + 0];
        float dq1 = q[i * QD + 1] - q[j * QD + 1];
        float dq2 = q[i * QD + 2] - q[j * QD + 2];
        float nd = sqrtf(dq0 * dq0 + dq1 * dq1 + dq2 * dq2);
        float invd = 1.0f / (nd + ESPV);
        g = invd * invd - invd;                     // dd^-2 - dd^-1
        msk = mvec[i] * mvec[j];
        u0 = dq0 * invd; u1 = dq1 * invd; u2 = dq2 * invd;  // i==j -> 0
      }
      gk[lane] = g; mk[lane] = msk;
    }
    LDS_FENCE();

    // layer1 collapsed: h1 = tanh(msk*(g*W1[0,:] + a_i + a_j))  (fp32 VALU)
    for (int idx = lane; idx < 16 * 128; idx += 32) {
      int p = idx >> 7;
      int h = idx & 127;
      int j = jn[p];
      float hv = 0.0f;
      if (j >= 0)
        hv = tanhf(mk[p] * (gk[p] * frW1[h] + a[i * H + h] + a[j * H + h]));
      h1[idx] = (h16)hv;
    }
    LDS_FENCE();

    // layer2: h2 = tanh(h1 @ W2)
    for (int nt = 0; nt < 8; ++nt) {
      v8f acc = {};
#pragma unroll
      for (int ks = 0; ks < 4; ++ks) {
        v16h A = wmma_load_A(h1, 128, ks * 32, lane);
        v16h B = wmma_load_B(W2T, 128, nt, ks * 32, lane);
        acc = wmma16(A, B, acc);
      }
#pragma unroll
      for (int r = 0; r < 8; ++r)
        h2[(r + 8 * hi) * 128 + nt * 16 + n_lane] = (h16)tanhf(acc[r]);
    }
    LDS_FENCE();

    // layer3: val = h2 @ W3
    {
      v8f acc = {};
#pragma unroll
      for (int ks = 0; ks < 4; ++ks) {
        v16h A = wmma_load_A(h2, 128, ks * 32, lane);
        v16h B = wmma_load_Bvec(W3h, ks * 32, lane);
        acc = wmma16(A, B, acc);
      }
      if (n_lane == 0) {
#pragma unroll
        for (int r = 0; r < 8; ++r) val16[r + 8 * hi] = acc[r];
      }
    }
    LDS_FENCE();

    if (lane < 16 && valid) {
      float vv = val16[lane];
      atomicAdd(&f[i * QD + 0], u0 * vv);
      atomicAdd(&f[i * QD + 1], u1 * vv);
      atomicAdd(&f[i * QD + 2], u2 * vv);
    }
    LDS_FENCE();
  }
}

extern "C" void kernel_launch(void* const* d_in, const int* in_sizes, int n_in,
                              void* d_out, int out_size, void* d_ws, size_t ws_size,
                              hipStream_t stream) {
  (void)in_sizes; (void)n_in; (void)out_size; (void)ws_size;
  const float* v    = (const float*)d_in[0];
  const float* e    = (const float*)d_in[1];
  const float* m_   = (const float*)d_in[2];
  const float* q    = (const float*)d_in[3];
  const float* vew1 = (const float*)d_in[4];
  const float* vew2 = (const float*)d_in[5];
  const float* mvw  = (const float*)d_in[6];
  const float* fbW1 = (const float*)d_in[7];
  const float* fbb1 = (const float*)d_in[8];
  const float* fbW2 = (const float*)d_in[9];
  const float* fbb2 = (const float*)d_in[10];
  const float* fbW3 = (const float*)d_in[11];
  const float* fbb3 = (const float*)d_in[12];
  const float* frW1 = (const float*)d_in[13];
  const float* frW2 = (const float*)d_in[14];
  const float* frW3 = (const float*)d_in[15];
  float* f = (float*)d_out;

  char* ws = (char*)d_ws;
  size_t off = 0;
  auto alloc = [&](size_t bytes) -> void* {
    void* p = ws + off;
    off = (off + bytes + 255) & ~(size_t)255;
    return p;
  };
  int*   srcI  = (int*)alloc(N_EDGE * 4);
  int*   dstI  = (int*)alloc(N_EDGE * 4);
  int*   molI  = (int*)alloc(N_NODE * 4);
  int*   gcnt  = (int*)alloc(M_MOL * 4);
  int*   glist = (int*)alloc((size_t)M_MOL * N_NODE * 4);
  float* aBuf  = (float*)alloc((size_t)N_NODE * H * 4);
  h16*   W1Tb  = (h16*)alloc(128 * 192 * 2);
  h16*   W2Tb  = (h16*)alloc(128 * 128 * 2);
  h16*   W3b   = (h16*)alloc(128 * 2);
  h16*   W2Tr  = (h16*)alloc(128 * 128 * 2);
  h16*   W3r   = (h16*)alloc(128 * 2);

  k_zero<<<(N_NODE * QD + 255) / 256, 256, 0, stream>>>(f, N_NODE * QD);
  k_decode<<<(N_NODE * N_EDGE + 255) / 256, 256, 0, stream>>>(vew1, srcI, 11, N_NODE * N_EDGE);
  k_decode<<<(N_NODE * N_EDGE + 255) / 256, 256, 0, stream>>>(vew2, dstI, 11, N_NODE * N_EDGE);
  k_decode<<<(M_MOL * N_NODE + 255) / 256, 256, 0, stream>>>(mvw, molI, 10, M_MOL * N_NODE);
  k_group<<<1, 32, 0, stream>>>(molI, glist, gcnt);
  k_transpose16<<<(192 * 128 + 255) / 256, 256, 0, stream>>>(fbW1, W1Tb, 192, 7);
  k_transpose16<<<(128 * 128 + 255) / 256, 256, 0, stream>>>(fbW2, W2Tb, 128, 7);
  k_transpose16<<<(128 * 128 + 255) / 256, 256, 0, stream>>>(frW2, W2Tr, 128, 7);
  k_cvt16<<<1, 128, 0, stream>>>(fbW3, W3b, 128);
  k_cvt16<<<1, 128, 0, stream>>>(frW3, W3r, 128);
  k_prec_a<<<N_NODE, H, 0, stream>>>(v, frW1, aBuf);
  k_bond<<<64, 128, 0, stream>>>(v, e, q, srcI, dstI, W1Tb, fbb1, W2Tb, fbb2, W3b, fbb3, f);
  k_pair<<<M_MOL, 128, 0, stream>>>(q, m_, aBuf, frW1, W2Tr, W3r, glist, gcnt, f);
}
